// QuotRemLinear_23210003268285
// MI455X (gfx1250) — matrix operand order repaired and checked by reading, not compiled, not run on hardware
//
#include <hip/hip_runtime.h>
#include <stdint.h>

// ---------------------------------------------------------------------------
// Problem constants (from reference)
// ---------------------------------------------------------------------------
constexpr int H_DIM   = 4096;     // hidden
constexpr int OUT_DIM = 4096;     // output features
constexpr int ROWS    = 4 * 4096; // B*S = 16384
constexpr int NGROUPS = 32;       // H / 128
constexpr int GS      = 128;      // group size (base and residual groups)

typedef __attribute__((ext_vector_type(16))) __bf16       v16bf;
typedef __attribute__((ext_vector_type(8)))  float        v8f;
typedef __attribute__((ext_vector_type(8)))  unsigned int v8u;

// Tensor Data Mover descriptor vectors (this toolchain: 6-arg builtin form)
#if defined(__AMDGCN__) && __has_builtin(__builtin_amdgcn_tensor_load_to_lds) && \
    __has_builtin(__builtin_amdgcn_s_wait_tensorcnt)
#define USE_TDM 1
typedef __attribute__((ext_vector_type(4))) unsigned int tdm_g0_t;
typedef __attribute__((ext_vector_type(8))) int          tdm_g1_t;
typedef __attribute__((ext_vector_type(4))) int          tdm_g2_t;
typedef __attribute__((ext_vector_type(8))) int          tdm_g4_t;
#endif

static __device__ __forceinline__ unsigned short f2bf(float f) {
  // round-to-nearest-even fp32 -> bf16
  unsigned int u = __float_as_uint(f);
  unsigned int r = u + 0x7FFFu + ((u >> 16) & 1u);
  return (unsigned short)(r >> 16);
}

// ---------------------------------------------------------------------------
// Kernel 1: fused quotient/residual fake-quant -> bf16 activation Y
// One wave32 per 128-channel group; 4 elements per lane.
// ---------------------------------------------------------------------------
__global__ __launch_bounds__(256) void quant_kernel(
    const float* __restrict__ x,
    const float* __restrict__ fixed_base,
    const float* __restrict__ fixed_sign,
    unsigned short* __restrict__ Y) {
  const int wave = threadIdx.x >> 5;
  const int lane = threadIdx.x & 31;
  const int gidx = blockIdx.x * 8 + wave;   // global group id
  const int row  = gidx >> 5;               // token row (0..16383)
  const int g    = gidx & 31;               // channel group (0..31)

  const float base = fixed_base[g];
  const float sign = fixed_sign[g];

  const size_t off = (size_t)row * H_DIM + g * GS + lane * 4;
  const float4 xv = *(const float4*)(x + off);
  float xe[4] = {xv.x, xv.y, xv.z, xv.w};

  float qv[4], rv[4];
  float mx = 0.0f;
#pragma unroll
  for (int i = 0; i < 4; ++i) {
    float q = (xe[i] * sign >= base * 0.5f) ? 1.0f : 0.0f;
    float r = xe[i] - sign * base * q;
    qv[i] = q; rv[i] = r;
    mx = fmaxf(mx, fabsf(r));
  }
  // wave32 max-reduction over the 128-element group
#pragma unroll
  for (int d = 16; d > 0; d >>= 1)
    mx = fmaxf(mx, __shfl_xor(mx, d, 32));

  const float scale = fmaxf(mx, 1e-8f) * (1.0f / 7.0f);
  const float inv   = 1.0f / scale;

  unsigned short o[4];
#pragma unroll
  for (int i = 0; i < 4; ++i) {
    float rq = fminf(fmaxf(rintf(rv[i] * inv), -8.0f), 7.0f) * scale;
    float y  = qv[i] * sign * base + rq;
    o[i] = f2bf(y);
  }
  uint2 pk;
  pk.x = (unsigned int)o[0] | ((unsigned int)o[1] << 16);
  pk.y = (unsigned int)o[2] | ((unsigned int)o[3] << 16);
  *(uint2*)(Y + off) = pk;
}

// ---------------------------------------------------------------------------
// Kernel 2: weight fp32 (OUT,H) -> bf16 TRANSPOSED Wt (H,OUT), 64x64 LDS tiles.
// k-major weight makes GEMM B staging a pure contiguous b128 copy.
// ---------------------------------------------------------------------------
__global__ __launch_bounds__(256) void wtrans_kernel(
    const float* __restrict__ w, unsigned short* __restrict__ wt) {
  __shared__ unsigned short t[64][64 + 2];
  const int o0 = blockIdx.x * 64;
  const int h0 = blockIdx.y * 64;
  const int tr = threadIdx.x >> 4;        // 0..15
  const int tc = (threadIdx.x & 15) * 4;  // 0..60
#pragma unroll
  for (int s = 0; s < 64; s += 16) {
    const float4 v = *(const float4*)(w + (size_t)(o0 + tr + s) * H_DIM + h0 + tc);
    t[tr + s][tc + 0] = f2bf(v.x);
    t[tr + s][tc + 1] = f2bf(v.y);
    t[tr + s][tc + 2] = f2bf(v.z);
    t[tr + s][tc + 3] = f2bf(v.w);
  }
  __syncthreads();
#pragma unroll
  for (int s = 0; s < 64; s += 16) {
    ushort4 ov;
    ov.x = t[tc + 0][tr + s];
    ov.y = t[tc + 1][tr + s];
    ov.z = t[tc + 2][tr + s];
    ov.w = t[tc + 3][tr + s];
    *(ushort4*)(wt + (size_t)(h0 + tr + s) * OUT_DIM + o0 + tc) = ov;
  }
}

// ---------------------------------------------------------------------------
// Kernel 3: bf16 WMMA GEMM  out[16384][4096] = Y @ Wt + bias
// Block: 256 threads (8 waves). Tile: BM=64 x BN=128, BK=32.
// Tiles staged to LDS by the Tensor Data Mover (wave 0 issues two D#s,
// pad_enable reproduces the +8-short row padding); manual b128 copy fallback.
// ---------------------------------------------------------------------------
constexpr int BM  = 64;
constexpr int BN  = 128;
constexpr int BK  = 32;
constexpr int LDA = BK + 8;   // 40 shorts = 16 data + 4 pad dwords per row
constexpr int LDB = BN + 8;   // 136 shorts = 64 data + 4 pad dwords per row

__global__ __launch_bounds__(256) void gemm_kernel(
    const unsigned short* __restrict__ Y,
    const unsigned short* __restrict__ Wt,
    const float* __restrict__ bias,
    float* __restrict__ out) {
  __shared__ unsigned short lA[BM * LDA];   // [64][40]  (m, k) row-major
  __shared__ unsigned short lB[BK * LDB];   // [32][136] (k, n) k-major

  const int tid  = threadIdx.x;
  const int lane = tid & 31;
  const int w    = tid >> 5;
  const int m0   = blockIdx.y * BM;
  const int n0   = blockIdx.x * BN;
  const int mt   = w & 3;          // wave's m sub-tile (0..3)
  const int ng   = (w >> 2) * 4;   // wave's first n sub-tile (0 or 4)

  // A-fragment addressing (ISA 16-bit A layout): lane = M row, halves split K
  const int am = lane & 15;
  const int ah = lane >> 4;
  const unsigned short* lArow = &lA[(mt * 16 + am) * LDA];
  // B-fragment addressing: lane = N col, halves split K (same K order as A)
  const unsigned short* lBrow = &lB[lane & 15];

#ifdef USE_TDM
  const unsigned lds_a =
      (unsigned)(uintptr_t)(__attribute__((address_space(3))) unsigned short*)&lA[0];
  const unsigned lds_b =
      (unsigned)(uintptr_t)(__attribute__((address_space(3))) unsigned short*)&lB[0];
  // group1: [17:16]=data_size(2B) [20]=pad_en [24:22]=pad_interval [31:25]=pad_amount
  // A tile: 32 k x 64 rows; row = 16 dwords -> interval code 3; pad 4 dwords -> 3
  const tdm_g1_t g1a = {
      (int)((1u << 16) | (1u << 20) | (3u << 22) | (3u << 25)),
      (int)((unsigned)H_DIM << 16),   // tensor_dim0 = H (lo16 @ bits 63:48)
      (int)((unsigned)ROWS << 16),    // tensor_dim0 hi=0 | tensor_dim1 = ROWS lo16
      (int)(32u << 16),               // tensor_dim1 hi=0 | tile_dim0 = 32 (k)
      64,                             // tile_dim1 = 64 (rows) | tile_dim2 = 0
      H_DIM,                          // tensor_dim0_stride lo32
      0, 0};
  // B tile: 128 n x 32 k rows; row = 64 dwords -> interval code 5; pad 4 -> 3
  const tdm_g1_t g1b = {
      (int)((1u << 16) | (1u << 20) | (5u << 22) | (3u << 25)),
      (int)((unsigned)OUT_DIM << 16), // tensor_dim0 = OUT
      (int)((unsigned)H_DIM << 16),   // tensor_dim1 = H
      (int)(128u << 16),              // tile_dim0 = 128 (n)
      32,                             // tile_dim1 = 32 (k rows)
      OUT_DIM,                        // tensor_dim0_stride lo32
      0, 0};
  const tdm_g2_t gz  = {0, 0, 0, 0};
  const tdm_g4_t gz8 = {0, 0, 0, 0, 0, 0, 0, 0};
#else
  // manual staging indices (fallback)
  const int arow = tid >> 2;           // 0..63
  const int acol = (tid & 3) * 8;      // 8 bf16 = 16B
  const int brow = tid >> 3;           // 0..31 (k row)
  const int bcol = (tid & 7) * 16;     // 16 bf16 = 32B
  const unsigned short* gA = Y  + (size_t)(m0 + arow) * H_DIM + acol;
  const unsigned short* gB = Wt + (size_t)brow * OUT_DIM + n0 + bcol;
#endif

  v8f acc[4] = {v8f{}, v8f{}, v8f{}, v8f{}};

  for (int kk = 0; kk < H_DIM; kk += BK) {
#ifdef USE_TDM
    if (w == 0) {
      // group0: count=1 | lds_addr | 57b global_addr | type=2
      unsigned long long gaA =
          (unsigned long long)(uintptr_t)Y + ((size_t)m0 * H_DIM + kk) * 2ull;
      unsigned long long gaB =
          (unsigned long long)(uintptr_t)Wt + ((size_t)kk * OUT_DIM + n0) * 2ull;
      tdm_g0_t g0a = {1u, lds_a, (unsigned)gaA,
                      (unsigned)(gaA >> 32) | (2u << 30)};
      tdm_g0_t g0b = {1u, lds_b, (unsigned)gaB,
                      (unsigned)(gaB >> 32) | (2u << 30)};
      __builtin_amdgcn_tensor_load_to_lds(g0a, g1a, gz, gz, gz8, 0);
      __builtin_amdgcn_tensor_load_to_lds(g0b, g1b, gz, gz, gz8, 0);
      __builtin_amdgcn_s_wait_tensorcnt(0);
    }
#else
    uint4 av = *(const uint4*)(gA + kk);
    *(uint4*)&lA[arow * LDA + acol] = av;
    const unsigned short* gb = gB + (size_t)kk * OUT_DIM;
    uint4 bv0 = *(const uint4*)gb;
    uint4 bv1 = *(const uint4*)(gb + 8);
    *(uint4*)&lB[brow * LDB + bcol] = bv0;
    *(uint4*)&lB[brow * LDB + bcol + 8] = bv1;
#endif
    __syncthreads();

    // ---- A fragment (shared by this wave's 4 tiles) ----
    union { v8u u; v16bf b; } af;
#pragma unroll
    for (int v = 0; v < 8; ++v) {
      int kb = ((v & 4) ? 16 : 0) + ah * 8 + (v & 3) * 2;
      af.u[v] = *(const unsigned int*)(lArow + kb);
    }

#pragma unroll
    for (int t = 0; t < 4; ++t) {
      union { v8u u; v16bf b; } bfr;
      const unsigned short* p = lBrow + (ng + t) * 16;
#pragma unroll
      for (int v = 0; v < 8; ++v) {
        int kb = ((v & 4) ? 16 : 0) + ah * 8 + (v & 3) * 2;
        bfr.u[v] = (unsigned int)p[kb * LDB] | ((unsigned int)p[(kb + 1) * LDB] << 16);
      }
      acc[t] = __builtin_amdgcn_wmma_f32_16x16x32_bf16(
          false, af.b, false, bfr.b, (short)0, acc[t], false, false);
    }
    __syncthreads();
  }

  // ---- epilogue: C/D layout -> lane = N col, VGPR j -> row j (+8 hi half)
  const int nl = lane & 15;
#pragma unroll
  for (int t = 0; t < 4; ++t) {
    const int col = n0 + (ng + t) * 16 + nl;
    const float bv = bias[col];
#pragma unroll
    for (int j = 0; j < 8; ++j) {
      const int r = m0 + mt * 16 + ah * 8 + j;
      out[(size_t)r * OUT_DIM + col] = acc[t][j] + bv;
    }
  }
}

// ---------------------------------------------------------------------------
// Host launch. Inputs (setup_inputs order):
//   0: x (B,S,H) f32  1: weight (OUT,H) f32  2: bias (OUT) f32
//   3: fixed_base (32) f32  4: fixed_sign_flag (32) f32
// Workspace: Y bf16 (128 MB) + Wt bf16 (32 MB) = 160 MB.
// ---------------------------------------------------------------------------
extern "C" void kernel_launch(void* const* d_in, const int* in_sizes, int n_in,
                              void* d_out, int out_size, void* d_ws, size_t ws_size,
                              hipStream_t stream) {
  const float* x          = (const float*)d_in[0];
  const float* weight     = (const float*)d_in[1];
  const float* bias       = (const float*)d_in[2];
  const float* fixed_base = (const float*)d_in[3];
  const float* fixed_sign = (const float*)d_in[4];
  float* out = (float*)d_out;

  unsigned short* Y  = (unsigned short*)d_ws;
  unsigned short* Wt = Y + (size_t)ROWS * H_DIM;

  // 1) activation quant -> bf16 Y
  quant_kernel<<<(ROWS * NGROUPS) / 8, 256, 0, stream>>>(x, fixed_base, fixed_sign, Y);
  // 2) weight -> bf16, transposed to (H, OUT)
  dim3 tgrid(OUT_DIM / 64, H_DIM / 64);
  wtrans_kernel<<<tgrid, 256, 0, stream>>>(weight, Wt);
  // 3) WMMA GEMM + bias
  dim3 grid(OUT_DIM / BN, ROWS / BM);
  gemm_kernel<<<grid, 256, 0, stream>>>(Y, Wt, bias, out);
}
